// FinalClassifier_17111149707365
// MI455X (gfx1250) — compile-verified, matrix operand order checked by reading, not run on hardware
//
#include <hip/hip_runtime.h>
#include <hip/hip_bf16.h>

typedef __attribute__((ext_vector_type(16))) __bf16 v16bf;
typedef __attribute__((ext_vector_type(8)))  float  v8f;

union Frag {
    v16bf v;
    uint4 q[2];
};

#define M_TILE 64
#define N_TILE 64
#define K_DIM  1024
#define K_STEP 32
#define NT     (K_DIM / K_STEP)   // 32 K-iterations
#define NC     1000
#define NC_PAD 1024
#define ASTRIDE 1032   // 1024 + 8 bf16 pad -> row stride 2064B, 16B aligned, conflict-free b128
#define BSTRIDE 40     // 32 + 8 bf16 pad  -> row stride 80B,   16B aligned, conflict-free b128

__device__ __forceinline__ unsigned pack_bf2(float a, float b) {
    union { __bf16 h[2]; unsigned u; } p;
    p.h[0] = (__bf16)a;
    p.h[1] = (__bf16)b;
    return p.u;
}

// CDNA5 async copy: global -> LDS directly, 16B per lane, tracked by ASYNCcnt.
__device__ __forceinline__ void async_load_b128(unsigned lds_addr, const void* gptr) {
    asm volatile("global_load_async_to_lds_b128 %0, %1, off"
                 :: "v"(lds_addr), "v"(gptr)
                 : "memory");
}
// Async loads complete in order (per wave), so waiting <=1 with two tiles in
// flight retires exactly the older tile while the newer keeps flying.
__device__ __forceinline__ void wait_async0() {
    asm volatile("s_wait_asynccnt 0x0" ::: "memory");
}
__device__ __forceinline__ void wait_async1() {
    asm volatile("s_wait_asynccnt 0x1" ::: "memory");
}

// ---------------------------------------------------------------------------
// Kernel 1: W [1024 x 1000] fp32 row-major  ->  Wt [1024 x 1024] bf16,
// Wt[c][k] = bf16(W[k][c]); columns >= 1000 zero-filled.
// ---------------------------------------------------------------------------
__global__ void wt_transpose_kernel(const float* __restrict__ W,
                                    __bf16* __restrict__ Wt) {
    const int c = blockIdx.x;                       // 0..1023 output row (W column)
    const int k = blockIdx.y * 256 + threadIdx.x;   // 0..1023
    float v = 0.0f;
    if (c < NC) v = W[(size_t)k * NC + c];
    Wt[(size_t)c * NC_PAD + k] = (__bf16)v;
}

// ---------------------------------------------------------------------------
// Kernel 2: fused GroupNorm + affine + GEMM (bf16 WMMA) + bias + Mish.
// Grid: (NC_PAD/N_TILE = 16, N/M_TILE = 2048), block = 256 threads (8 waves).
// B tiles triple-buffered in LDS, distance-2 async prefetch (ASYNCcnt <= 1).
// ---------------------------------------------------------------------------
__global__ __launch_bounds__(256)
void gn_gemm_mish_kernel(const float* __restrict__ x,
                         const float* __restrict__ gamma,
                         const float* __restrict__ beta,
                         const __bf16* __restrict__ Wt,
                         const float* __restrict__ bias,
                         float* __restrict__ out) {
    __shared__ __align__(16) __bf16 Abuf[M_TILE * ASTRIDE];     // 132096 B
    __shared__ __align__(16) __bf16 Bbuf[3][N_TILE * BSTRIDE];  //  15360 B
    __shared__ float sgamma[K_DIM];                             //   4096 B
    __shared__ float sbeta[K_DIM];                              //   4096 B

    const int tid     = threadIdx.x;
    const int rowBase = blockIdx.y * M_TILE;   // global row base
    const int colBase = blockIdx.x * N_TILE;   // global (padded) col base

    // ---- B-tile async staging setup: each thread owns one b128 per tile ----
    const int brow = tid >> 2;                 // 0..63 (output column within tile)
    const int bseg = tid & 3;                  // K segment * 8
    const __bf16* gsrc = Wt + (size_t)(colBase + brow) * NC_PAD + (bseg << 3);
    const unsigned ldsB[3] = {
        (unsigned)(uintptr_t)&Bbuf[0][brow * BSTRIDE + (bseg << 3)],
        (unsigned)(uintptr_t)&Bbuf[1][brow * BSTRIDE + (bseg << 3)],
        (unsigned)(uintptr_t)&Bbuf[2][brow * BSTRIDE + (bseg << 3)]
    };

    // Kick off tile 0 now: its latency hides under the GroupNorm phase.
    async_load_b128(ldsB[0], gsrc);

    // ---- stage gamma/beta ----
    for (int i = tid; i < K_DIM; i += 256) {
        sgamma[i] = gamma[i];
        sbeta[i]  = beta[i];
    }
    __syncthreads();

    // ---- Phase 1: GroupNorm 64 rows x 1024 ch -> bf16 A-tile in LDS ----
    // group = 16 contiguous channels; thread handles 16 groups of one row.
    {
        const int row = tid >> 2;            // 0..63
        const int gq  = tid & 3;             // group sub-index
        const float* xrow = x + (size_t)(rowBase + row) * K_DIM;
        for (int gi = 0; gi < 16; ++gi) {
            const int g  = (gi << 2) + gq;   // 0..63
            const int ch = g << 4;           // first channel of group
            const float4* px = (const float4*)(xrow + ch);
            float4 f0 = px[0], f1 = px[1], f2 = px[2], f3 = px[3];
            float v[16] = { f0.x, f0.y, f0.z, f0.w, f1.x, f1.y, f1.z, f1.w,
                            f2.x, f2.y, f2.z, f2.w, f3.x, f3.y, f3.z, f3.w };
            float s = 0.f;
            #pragma unroll
            for (int c = 0; c < 16; ++c) s += v[c];
            const float mean = s * 0.0625f;
            float s2 = 0.f;
            #pragma unroll
            for (int c = 0; c < 16; ++c) {
                float d = v[c] - mean;
                s2 += d * d;
            }
            const float rstd = rsqrtf(s2 * 0.0625f + 1e-5f);

            union { unsigned u[8]; uint4 q[2]; } o;
            #pragma unroll
            for (int c = 0; c < 16; c += 2) {
                float y0 = (v[c]     - mean) * rstd * sgamma[ch + c]     + sbeta[ch + c];
                float y1 = (v[c + 1] - mean) * rstd * sgamma[ch + c + 1] + sbeta[ch + c + 1];
                o.u[c >> 1] = pack_bf2(y0, y1);
            }
            uint4* dst = (uint4*)&Abuf[row * ASTRIDE + ch];
            dst[0] = o.q[0];
            dst[1] = o.q[1];
        }
    }

    // Second tile in flight before the main loop (distance-2 pipeline).
    async_load_b128(ldsB[1], gsrc + K_STEP);

    // ---- Phase 2: GEMM over K with bf16 WMMA, triple-buffered B ----
    const int wave = tid >> 5;            // 0..7
    const int lane = tid & 31;
    const int l16  = lane & 15;
    const int lhi  = lane >> 4;           // 0 or 1
    const int cSub = wave & 3;            // col subtile 0..3
    const int rP   = (wave >> 2) << 1;    // row subtile pair base: 0 or 2

    v8f acc0 = {};
    v8f acc1 = {};

    for (int it = 0; it < NT; ++it) {
        const int kk = it * K_STEP;

        // Retire tile `it` (keep tile it+1 in flight), publish to workgroup.
        if (it < NT - 1) wait_async1(); else wait_async0();
        __syncthreads();

        // Prefetch tile it+2 into the buffer consumed at iteration it-1.
        // Safe: the barrier above ordered all reads of that buffer.
        if (it + 2 < NT)
            async_load_b128(ldsB[(it + 2) % 3], gsrc + kk + 2 * K_STEP);

        // B fragment (lane = output column, two 16B K-chunks at +0 / +32B)
        Frag bf_;
        {
            const uint4* p = (const uint4*)&Bbuf[it % 3]
                                 [((cSub << 4) + l16) * BSTRIDE + (lhi << 3)];
            bf_.q[0] = p[0];
            bf_.q[1] = p[2];
        }
        // A fragments for the two row subtiles
        Frag a0, a1;
        {
            const uint4* p0 = (const uint4*)&Abuf[((rP << 4) + l16) * ASTRIDE + kk + (lhi << 3)];
            a0.q[0] = p0[0];
            a0.q[1] = p0[2];
            const uint4* p1 = (const uint4*)&Abuf[(((rP + 1) << 4) + l16) * ASTRIDE + kk + (lhi << 3)];
            a1.q[0] = p1[0];
            a1.q[1] = p1[2];
        }

        acc0 = __builtin_amdgcn_wmma_f32_16x16x32_bf16(false, a0.v, false, bf_.v,
                                                       (short)0, acc0, false, false);
        acc1 = __builtin_amdgcn_wmma_f32_16x16x32_bf16(false, a1.v, false, bf_.v,
                                                       (short)0, acc1, false, false);
    }

    // ---- Epilogue: bias + Mish + store (C/D layout: vgpr i -> M=i+8*lhi, N=l16) ----
    const int n = colBase + (cSub << 4) + l16;
    if (n < NC) {
        const float bn = bias[n];
        #pragma unroll
        for (int j = 0; j < 2; ++j) {
            const v8f acc = j ? acc1 : acc0;
            const int m0 = rowBase + ((rP + j) << 4) + (lhi << 3);
            #pragma unroll
            for (int i = 0; i < 8; ++i) {
                float y = acc[i] + bn;
                // mish(y) = y * tanh(softplus(y)) = y * ((1+e^y)^2 - 1)/((1+e^y)^2 + 1)
                float t = __expf(fminf(y, 20.0f));
                float u = 1.0f + t;
                u *= u;
                out[(size_t)(m0 + i) * NC + n] = y * (u - 1.0f) / (u + 1.0f);
            }
        }
    }
}

extern "C" void kernel_launch(void* const* d_in, const int* in_sizes, int n_in,
                              void* d_out, int out_size, void* d_ws, size_t ws_size,
                              hipStream_t stream) {
    const float* x     = (const float*)d_in[0];
    const float* gamma = (const float*)d_in[1];
    const float* beta  = (const float*)d_in[2];
    const float* W     = (const float*)d_in[3];
    const float* b     = (const float*)d_in[4];
    float* out         = (float*)d_out;

    __bf16* Wt = (__bf16*)d_ws;   // 1024 x 1024 bf16 = 2 MB

    // 1) transpose + convert W -> Wt (bf16, zero-padded to 1024 cols)
    wt_transpose_kernel<<<dim3(NC_PAD, 4), 256, 0, stream>>>(W, Wt);

    // 2) fused GroupNorm + GEMM + Mish
    const int N = in_sizes[0] / K_DIM;    // 131072
    dim3 grid(NC_PAD / N_TILE, N / M_TILE);
    gn_gemm_mish_kernel<<<grid, 256, 0, stream>>>(x, gamma, beta, Wt, b, out);
}
